// GATv2Model_70042326663592
// MI455X (gfx1250) — matrix operand order repaired and checked by reading, not compiled, not run on hardware
//
#include <hip/hip_runtime.h>

#define NN 50000
#define EE 800000

typedef __attribute__((ext_vector_type(16))) __bf16 v16bf;
typedef __attribute__((ext_vector_type(8)))  __bf16 v8bf;
typedef __attribute__((ext_vector_type(4)))  __bf16 v4bf;
typedef __attribute__((ext_vector_type(8)))  float  v8f;

// ---------- helpers ----------
__device__ __forceinline__ __bf16 f2bf(float f) {
  // round-to-nearest-even f32 -> bf16
  unsigned u = __builtin_bit_cast(unsigned, f);
  unsigned r = (u + 0x7FFFu + ((u >> 16) & 1u)) >> 16;
  unsigned short s = (unsigned short)r;
  return __builtin_bit_cast(__bf16, s);
}

// order-preserving float <-> uint key (atomic max over possibly-negative floats)
__device__ __forceinline__ unsigned fkey(float f) {
  unsigned u = __builtin_bit_cast(unsigned, f);
  return (u & 0x80000000u) ? ~u : (u | 0x80000000u);
}
__device__ __forceinline__ float fdec(unsigned k) {
  unsigned u = (k & 0x80000000u) ? (k & 0x7FFFFFFFu) : ~k;
  return __builtin_bit_cast(float, u);
}

// two contiguous b128 loads -> one 16-element bf16 fragment
__device__ __forceinline__ v16bf load_frag(const __bf16* p0, const __bf16* p1) {
  v8bf lo = *(const v8bf*)p0;
  v8bf hi = *(const v8bf*)p1;
  return __builtin_shufflevector(lo, hi, 0, 1, 2, 3, 4, 5, 6, 7,
                                 8, 9, 10, 11, 12, 13, 14, 15);
}

// ---------- one-time converts / packs (per launch; tiny vs HBM bandwidth) ----------

// f32 -> bf16, 4 elements/thread, b128 in / b64 out
__global__ void f32_to_bf16(const float* __restrict__ in, __bf16* __restrict__ out, int n) {
  int i = (blockIdx.x * blockDim.x + threadIdx.x) * 4;
  if (i >= n) return;
  float4 f = *(const float4*)(in + i);
  v4bf o;
  o[0] = f2bf(f.x); o[1] = f2bf(f.y); o[2] = f2bf(f.z); o[3] = f2bf(f.w);
  *(v4bf*)(out + i) = o;
}

// Pack W[K x M] f32 into bf16 B-fragments: P[((mt*KC + kc)*32 + lane)*16 + j]
// holds W[(kc*32 + (lane>=16?8:0) + j + (j>=8?8:0)) * M + mt*16 + (lane&15)]
template<int K, int M>
__global__ void pack_w_bf16(const float* __restrict__ W, __bf16* __restrict__ P) {
  constexpr int KC = K / 32, MT = M / 16;
  int idx = blockIdx.x * blockDim.x + threadIdx.x;   // one thread per (mt,kc,lane)
  if (idx >= MT * KC * 32) return;
  int lane = idx & 31;
  int t    = idx >> 5;
  int kc   = t % KC;
  int mt   = t / KC;
  int ncol  = (mt << 4) + (lane & 15);
  int kbase = (kc << 5) + ((lane >> 4) << 3);
  __bf16* dst = P + (size_t)idx * 16;
#pragma unroll
  for (int j = 0; j < 16; ++j)
    dst[j] = f2bf(W[(size_t)(kbase + j + (j >= 8 ? 8 : 0)) * M + ncol]);
}

// ---------- dense transform: Y[n x M] = Xbf[n x K] @ W + bias ----------
// One wave computes a full 16-row x M strip. A/B fragments are double-buffered:
// the loads for fragment i+1 issue before the WMMA for fragment i, so waits are
// partial instead of drain-to-zero. K,M compile-time -> fully unrolled pipeline.
template<int K, int M>
__global__ void __launch_bounds__(128)
wmma_gemm_pk(const __bf16* __restrict__ Xbf, const __bf16* __restrict__ Wp,
             const float* __restrict__ bias, float* __restrict__ Y, int n) {
  constexpr int KC = K / 32, MT = M / 16;
  const int wave  = threadIdx.x >> 5;
  const int lane  = threadIdx.x & 31;
  const int strip = blockIdx.x * 4 + wave;           // 16-row strip index
  if (strip >= (n >> 4)) return;                     // wave-uniform guard
  const int row0 = strip << 4;
  const int l15  = lane & 15;
  const int hi   = lane >> 4;
  const int khA  = hi << 3;                          // A: K offset 0/8 within 16-chunk

  v8f acc[MT];
#pragma unroll
  for (int mt = 0; mt < MT; ++mt) {
    float bv = bias ? bias[(mt << 4) + l15] : 0.0f;
#pragma unroll
    for (int r = 0; r < 8; ++r) acc[mt][r] = bv;
  }

  const __bf16* xrow = Xbf + (size_t)(row0 + l15) * K;
  const __bf16* wbase = Wp + (size_t)lane * 16;

  // flattened (kc, mt) pipeline with 2-deep A and B fragment buffers
  v16bf abuf[2], bbuf[2];
  abuf[0] = load_frag(xrow + khA, xrow + 16 + khA);
  bbuf[0] = load_frag(wbase, wbase + 8);             // frag (kc=0, mt=0)

  constexpr int TOT = KC * MT;
#pragma unroll
  for (int i = 0; i < TOT; ++i) {
    const int kc = i / MT, mt = i % MT;
    const int ni = i + 1;
    if (ni < TOT) {
      const int nkc = ni / MT, nmt = ni % MT;
      const __bf16* bp = wbase + (size_t)(nmt * KC + nkc) * 32 * 16;
      bbuf[ni & 1] = load_frag(bp, bp + 8);
      if (nmt == 0)
        abuf[nkc & 1] = load_frag(xrow + (nkc << 5) + khA,
                                  xrow + (nkc << 5) + 16 + khA);
    }
    acc[mt] = __builtin_amdgcn_wmma_f32_16x16x32_bf16(
        false, abuf[kc & 1], false, bbuf[i & 1], (short)0, acc[mt], false, false);
  }

  // C/D: lanes 0-15 -> N=lane, VGPR r -> M=r ; lanes 16-31 -> N=lane-16, M=8+r
  const int m0 = hi << 3;
#pragma unroll
  for (int mt = 0; mt < MT; ++mt) {
    float* yp = Y + (mt << 4) + l15;
#pragma unroll
    for (int r = 0; r < 8; ++r)
      yp[(size_t)(row0 + m0 + r) * M] = acc[mt][r];
  }
}

// ---------- edge phase ----------
__global__ void init_softmax(unsigned* __restrict__ mkey, float* __restrict__ denom, int n) {
  int i = blockIdx.x * blockDim.x + threadIdx.x;
  if (i < n) { mkey[i] = 0x007FFFFFu; denom[i] = 0.0f; }   // key(-inf), 0
}

__global__ void zero_f32(float* __restrict__ p, int n) {
  int i = blockIdx.x * blockDim.x + threadIdx.x;
  if (i < n) p[i] = 0.0f;
}

// pass 1: logit[e,h] = att[h,:] . leaky_relu(xl[src]+xr[dst]) ; segment max via u32 atomic
__global__ void edge_logit_max(const float* __restrict__ xl, const float* __restrict__ xr,
                               const float* __restrict__ att,
                               const int* __restrict__ src, const int* __restrict__ dst,
                               float* __restrict__ lp, unsigned* __restrict__ mkey,
                               int ne, int H, int C) {
  int i = blockIdx.x * blockDim.x + threadIdx.x;
  if (i >= ne * H) return;
  int e = i / H, h = i % H;
  int s = src[e], d = dst[e];
  int HC = H * C;
  const float* a = xl + (size_t)s * HC + h * C;
  const float* b = xr + (size_t)d * HC + h * C;
  const float* w = att + h * C;
  float acc = 0.0f;
  for (int c = 0; c < C; c += 4) {
    float4 av = *(const float4*)(a + c);
    float4 bv = *(const float4*)(b + c);
    float4 wv = *(const float4*)(w + c);
    float v0 = av.x + bv.x, v1 = av.y + bv.y, v2 = av.z + bv.z, v3 = av.w + bv.w;
    v0 = v0 > 0.f ? v0 : v0 * 0.2f;
    v1 = v1 > 0.f ? v1 : v1 * 0.2f;
    v2 = v2 > 0.f ? v2 : v2 * 0.2f;
    v3 = v3 > 0.f ? v3 : v3 * 0.2f;
    acc += v0 * wv.x + v1 * wv.y + v2 * wv.z + v3 * wv.w;
  }
  lp[i] = acc;
  atomicMax(&mkey[(size_t)d * H + h], fkey(acc));
}

// pass 2: p = exp(logit - m); denom[dst,h] += p  (p overwrites logit in place)
__global__ void edge_exp(const int* __restrict__ dst, float* __restrict__ lp,
                         const unsigned* __restrict__ mkey, float* __restrict__ denom,
                         int ne, int H) {
  int i = blockIdx.x * blockDim.x + threadIdx.x;
  if (i >= ne * H) return;
  int e = i / H, h = i % H;
  int d = dst[e];
  float m = fdec(mkey[(size_t)d * H + h]);
  float p = __expf(lp[i] - m);
  lp[i] = p;
  atomicAdd(&denom[(size_t)d * H + h], p);
}

// pass 3: agg[dst, c] += xl[src, c] * p/(denom+eps)   (one thread per edge-channel)
__global__ void edge_aggregate(const float* __restrict__ xl,
                               const int* __restrict__ src, const int* __restrict__ dst,
                               const float* __restrict__ p, const float* __restrict__ denom,
                               float* __restrict__ agg, int ne, int H, int C) {
  int HC = H * C;
  long long i = (long long)blockIdx.x * blockDim.x + threadIdx.x;
  if (i >= (long long)ne * HC) return;
  int e = (int)(i / HC), c = (int)(i % HC);
  int h = c / C;
  int s = src[e], d = dst[e];
  float alpha = p[(size_t)e * H + h] / (denom[(size_t)d * H + h] + 1e-16f);
  atomicAdd(&agg[(size_t)d * HC + c], xl[(size_t)s * HC + c] * alpha);
}

// ---------- finalize ----------
__global__ void elu_kernel(const float* __restrict__ agg, float* __restrict__ h, int n) {
  int i = blockIdx.x * blockDim.x + threadIdx.x;
  if (i >= n) return;
  float a = agg[i];
  h[i] = a > 0.0f ? a : expm1f(a);
}

__global__ void bias_out(const float* __restrict__ agg, const float* __restrict__ b,
                         float* __restrict__ out, int n, int M) {
  int i = blockIdx.x * blockDim.x + threadIdx.x;
  if (i >= n) return;
  out[i] = agg[i] + b[i % M];
}

// ---------- host orchestration ----------
struct LayerBufs {
  __bf16* xbf;                // input activations, bf16 (N x 128)
  __bf16 *wpl, *wpr, *wpres;  // packed weights
  float *xl, *xr, *agg;
  unsigned* mkey;
  float *denom, *lp;
};

template<int M>
static void run_gat_layer(const float* xin,
                          const float* Wl, const float* bl,
                          const float* Wr, const float* br,
                          const float* att, const float* resW, const float* lbias,
                          const int* src, const int* dst,
                          const LayerBufs& B, int H, int C, hipStream_t stream) {
  constexpr int K = 128;
  // convert activations once; pack weights (tiny)
  f32_to_bf16<<<(NN * K / 4 + 255) / 256, 256, 0, stream>>>(xin, B.xbf, NN * K);
  constexpr int packThreads = (M / 16) * (K / 32) * 32;
  pack_w_bf16<K, M><<<(packThreads + 255) / 256, 256, 0, stream>>>(Wl, B.wpl);
  pack_w_bf16<K, M><<<(packThreads + 255) / 256, 256, 0, stream>>>(Wr, B.wpr);
  if (resW) pack_w_bf16<K, M><<<(packThreads + 255) / 256, 256, 0, stream>>>(resW, B.wpres);

  const int gblocks = (NN / 16 + 3) / 4;               // 4 waves/block, 1 strip/wave
  wmma_gemm_pk<K, M><<<gblocks, 128, 0, stream>>>(B.xbf, B.wpl, bl, B.xl, NN);
  wmma_gemm_pk<K, M><<<gblocks, 128, 0, stream>>>(B.xbf, B.wpr, br, B.xr, NN);
  if (resW)  // residual + layer bias seed the aggregation buffer directly
    wmma_gemm_pk<K, M><<<gblocks, 128, 0, stream>>>(B.xbf, B.wpres, lbias, B.agg, NN);
  else
    zero_f32<<<(NN * M + 255) / 256, 256, 0, stream>>>(B.agg, NN * M);

  init_softmax<<<(NN * H + 255) / 256, 256, 0, stream>>>(B.mkey, B.denom, NN * H);
  edge_logit_max<<<(EE * H + 255) / 256, 256, 0, stream>>>(B.xl, B.xr, att, src, dst,
                                                           B.lp, B.mkey, EE, H, C);
  edge_exp<<<(EE * H + 255) / 256, 256, 0, stream>>>(dst, B.lp, B.mkey, B.denom, EE, H);
  unsigned agblocks = (unsigned)(((long long)EE * M + 255) / 256);
  edge_aggregate<<<agblocks, 256, 0, stream>>>(B.xl, src, dst, B.lp, B.denom, B.agg, EE, H, C);
}

extern "C" void kernel_launch(void* const* d_in, const int* in_sizes, int n_in,
                              void* d_out, int out_size, void* d_ws, size_t ws_size,
                              hipStream_t stream) {
  const float* x    = (const float*)d_in[0];
  const int*   ei   = (const int*)d_in[1];
  const int*   src  = ei;
  const int*   dst  = ei + EE;
  const float* Wl0  = (const float*)d_in[2];
  const float* bl0  = (const float*)d_in[3];
  const float* Wr0  = (const float*)d_in[4];
  const float* br0  = (const float*)d_in[5];
  const float* att0 = (const float*)d_in[6];
  const float* b0   = (const float*)d_in[7];
  const float* res0 = (const float*)d_in[8];
  const float* Wl1  = (const float*)d_in[9];
  const float* bl1  = (const float*)d_in[10];
  const float* Wr1  = (const float*)d_in[11];
  const float* br1  = (const float*)d_in[12];
  const float* att1 = (const float*)d_in[13];
  const float* b1   = (const float*)d_in[14];
  const float* res1 = (const float*)d_in[15];
  const float* Wl2  = (const float*)d_in[16];
  const float* bl2  = (const float*)d_in[17];
  const float* Wr2  = (const float*)d_in[18];
  const float* br2  = (const float*)d_in[19];
  const float* att2 = (const float*)d_in[20];
  const float* b2   = (const float*)d_in[21];

  // workspace layout (floats): xl | xr | agg | hbuf | mkey | denom | lp | xbf | wpacks
  float* ws = (float*)d_ws;
  const size_t NH = (size_t)NN * 128;
  float*    xl    = ws;
  float*    xr    = xl + NH;
  float*    agg   = xr + NH;
  float*    hbuf  = agg + NH;
  unsigned* mkey  = (unsigned*)(hbuf + NH);
  float*    denom = (float*)(mkey + (size_t)NN * 4);
  float*    lp    = denom + (size_t)NN * 4;
  __bf16*   xbf   = (__bf16*)(lp + (size_t)EE * 4);
  __bf16*   wpl   = xbf + NH;                // 128x128 bf16 each
  __bf16*   wpr   = wpl + 128 * 128;
  __bf16*   wpres = wpr + 128 * 128;

  LayerBufs B{xbf, wpl, wpr, wpres, xl, xr, agg, mkey, denom, lp};

  // layer 0: 128 -> 4x32 concat, residual, ELU
  run_gat_layer<128>(x, Wl0, bl0, Wr0, br0, att0, res0, b0, src, dst, B, 4, 32, stream);
  elu_kernel<<<(NN * 128 + 255) / 256, 256, 0, stream>>>(agg, hbuf, NN * 128);

  // layer 1: 128 -> 4x32 concat, residual, ELU (in-place over hbuf)
  run_gat_layer<128>(hbuf, Wl1, bl1, Wr1, br1, att1, res1, b1, src, dst, B, 4, 32, stream);
  elu_kernel<<<(NN * 128 + 255) / 256, 256, 0, stream>>>(agg, hbuf, NN * 128);

  // layer 2: 128 -> 64, single head, no residual, no activation
  run_gat_layer<64>(hbuf, Wl2, bl2, Wr2, br2, att2, nullptr, nullptr, src, dst, B, 1, 64, stream);
  bias_out<<<(NN * 64 + 255) / 256, 256, 0, stream>>>(agg, b2, (float*)d_out, NN * 64, 64);
}
